// Gnn_basic_17978733101277
// MI455X (gfx1250) — matrix operand order repaired and checked by reading, not compiled, not run on hardware
//
#include <hip/hip_runtime.h>
#include <hip/hip_bf16.h>
#include <cstdint>

#define WIDTH 128
#define BATCH 2
#define NNODE 10000
#define NEDGE 160000

typedef __attribute__((ext_vector_type(16))) __bf16 v16bf;
typedef __attribute__((ext_vector_type(8)))  __bf16 v8bf;
typedef __attribute__((ext_vector_type(8)))  float  v8f;

struct MlpW { const unsigned short* Wt[6]; const float* b[6]; };

__device__ __forceinline__ unsigned short f2bf(float f) {
  union { float f; unsigned u; } c; c.f = f;
  unsigned u = c.u;
  u += 0x7fffu + ((u >> 16) & 1u);          // round-to-nearest-even
  return (unsigned short)(u >> 16);
}
__device__ __forceinline__ float bf2f(unsigned short h) {
  union { unsigned u; float f; } c; c.u = ((unsigned)h) << 16; return c.f;
}

// ---- WMMA GEMM over one wave's 16 rows, NT 16-col output tiles ----
// A layout (16x32 bf16): lanes 0-15 M=0..15 K=[0..7,16..23]; lanes 16-31 K=[8..15,24..31]
// B layout (32x16 bf16): lane n=l&15, g=l>>4 holds K=[kc+8g..+7, kc+16+8g..+7] (Wt is [N][K])
template<int NT>
__device__ __forceinline__ void gemm_rows16(
    const unsigned short* __restrict__ Wt, int K,
    const float* __restrict__ bias, int biasN,
    const unsigned short* in, int sIn, int rowBase,
    v8f (&acc)[NT])
{
  const int lane = threadIdx.x & 31;
  const int g    = lane >> 4;
  const int lm   = lane & 15;
  for (int t = 0; t < NT; ++t) {
    int col = t * 16 + lm;
    float bv = (col < biasN) ? bias[col] : 0.f;
    v8f z;
    for (int r = 0; r < 8; ++r) z[r] = bv;
    acc[t] = z;
  }
  const int arow = rowBase + lm;
  for (int kc = 0; kc < K; kc += 32) {
    union { v16bf v; struct { v8bf lo, hi; } p; } a;
    a.p.lo = *(const v8bf*)(in + arow * sIn + kc + 8 * g);
    a.p.hi = *(const v8bf*)(in + arow * sIn + kc + 16 + 8 * g);
#pragma unroll
    for (int t = 0; t < NT; ++t) {
      int n = t * 16 + lm;
      union { v16bf v; struct { v8bf lo, hi; } p; } bu;
      bu.p.lo = *(const v8bf*)(Wt + (size_t)n * K + kc + 8 * g);
      bu.p.hi = *(const v8bf*)(Wt + (size_t)n * K + kc + 16 + 8 * g);
      v16bf bb = bu.v;
      acc[t] = __builtin_amdgcn_wmma_f32_16x16x32_bf16(
          false, a.v, false, bb, (short)0, acc[t], false, false);
    }
  }
}

// D layout (16x16 f32): VGPR r, lane l -> row r + 8*(l>>4), col l&15
__device__ __forceinline__ void store_tile(
    unsigned short* out, int sOut, int rowBase, int t, const v8f& a,
    const unsigned short* res, int sRes, bool relu)
{
  const int lane = threadIdx.x & 31, g = lane >> 4, lm = lane & 15;
  const int col = t * 16 + lm;
#pragma unroll
  for (int r = 0; r < 8; ++r) {
    int row = rowBase + 8 * g + r;
    float v = a[r];
    if (relu && v < 0.f) v = 0.f;
    if (res) v += bf2f(res[row * sRes + col]);
    out[row * sOut + col] = f2bf(v);
  }
}

__device__ __forceinline__ void layer128(
    const unsigned short* __restrict__ Wt, const float* __restrict__ bias,
    const unsigned short* in, int sIn, int K,
    unsigned short* out, int sOut,
    const unsigned short* res, int sRes, bool relu)
{
  const int rowBase = (threadIdx.x >> 5) * 16;
  v8f acc[8];
  gemm_rows16<8>(Wt, K, bias, WIDTH, in, sIn, rowBase, acc);
#pragma unroll
  for (int t = 0; t < 8; ++t) store_tile(out, sOut, rowBase, t, acc[t], res, sRes, relu);
  __syncthreads();
}

// ---- fp32 weight [K][N] -> bf16 transposed+padded [Np][Kp] ----
__global__ void convW(const float* __restrict__ W, unsigned short* __restrict__ Wt,
                      int K, int Kp, int N, int Np)
{
  int i = blockIdx.x * blockDim.x + threadIdx.x;
  if (i >= Np * Kp) return;
  int n = i / Kp, k = i - n * Kp;
  unsigned short v = 0;
  if (k < K && n < N) v = f2bf(W[(size_t)k * N + n]);
  Wt[i] = v;
}

// ---- Kernel 1: node_encoder  nf[B,N,8] -> ne[B,N,128] bf16 ----
__global__ void __launch_bounds__(128) node_encoder_kernel(
    const float* __restrict__ nf, unsigned short* __restrict__ ne,
    MlpW w, int Nn)
{
  __shared__ alignas(16) unsigned short bufA[64 * 136];
  __shared__ alignas(16) unsigned short bufB[64 * 136];
  const int b = blockIdx.y, tid = threadIdx.x;
  const int tileBase = blockIdx.x * 64;
  {
    int row = tid >> 1, kb = (tid & 1) * 16, gr = tileBase + row;
    for (int k = 0; k < 16; ++k) {
      int kk = kb + k; unsigned short v = 0;
      if (kk < 8 && gr < Nn) v = f2bf(nf[((size_t)b * Nn + gr) * 8 + kk]);
      bufB[row * 32 + kk] = v;
    }
  }
  __syncthreads();
  layer128(w.Wt[0], w.b[0], bufB, 32, 32,   bufA, 136, nullptr, 0, true);
  layer128(w.Wt[1], w.b[1], bufA, 136, 128, bufB, 136, nullptr, 0, true);
  layer128(w.Wt[2], w.b[2], bufB, 136, 128, bufA, 136, bufA, 136, true);
  layer128(w.Wt[3], w.b[3], bufA, 136, 128, bufB, 136, nullptr, 0, true);
  layer128(w.Wt[4], w.b[4], bufB, 136, 128, bufA, 136, bufA, 136, true);
  layer128(w.Wt[5], w.b[5], bufA, 136, 128, bufB, 136, nullptr, 0, false);
  {
    int row = tid >> 1, cb = (tid & 1) * 64, gr = tileBase + row;
    if (gr < Nn) {
      unsigned short* dst = ne + ((size_t)b * Nn + gr) * WIDTH + cb;
      const unsigned short* src = bufB + row * 136 + cb;
#pragma unroll
      for (int i = 0; i < 8; ++i) *(uint4*)(dst + i * 8) = *(const uint4*)(src + i * 8);
    }
  }
}

// ---- Kernel 2: fused edge_encoder + gather + edge_processor + atomic segment-sum ----
__global__ void __launch_bounds__(64) edge_kernel(
    const int* __restrict__ edges, const float* __restrict__ ef,
    const unsigned short* __restrict__ ne, float* __restrict__ agg,
    MlpW we, MlpW wp, int Nn, int Ee)
{
  __shared__ alignas(16) unsigned short bufA[32 * 392];   // 384-wide concat buffer
  __shared__ alignas(16) unsigned short bufB[32 * 136];
  __shared__ int sIdx[32];
  const int b = blockIdx.y, tid = threadIdx.x;
  const int tileBase = blockIdx.x * 32;
  unsigned short* r0 = bufA;         // ee
  unsigned short* r1 = bufA + 128;   // ne[src] (also ping buffer)
  unsigned short* r2 = bufA + 256;   // ne[dst]
  {
    int row = tid >> 1, kb = (tid & 1) * 16, e = tileBase + row;
    for (int k = 0; k < 16; ++k) {
      int kk = kb + k; unsigned short v = 0;
      if (kk < 4) v = f2bf(ef[((size_t)b * Ee + e) * 4 + kk]);
      bufB[row * 32 + kk] = v;
    }
    if (tid < 32) sIdx[tid] = edges[(size_t)b * 2 * Ee + tileBase + tid];
  }
  __syncthreads();
  // edge_encoder (h ping-pongs between r1 and bufB; ends in r0)
  layer128(we.Wt[0], we.b[0], bufB, 32, 32,   r1, 392, nullptr, 0, true);
  layer128(we.Wt[1], we.b[1], r1, 392, 128,   bufB, 136, nullptr, 0, true);
  layer128(we.Wt[2], we.b[2], bufB, 136, 128, r1, 392, r1, 392, true);
  layer128(we.Wt[3], we.b[3], r1, 392, 128,   bufB, 136, nullptr, 0, true);
  layer128(we.Wt[4], we.b[4], bufB, 136, 128, r1, 392, r1, 392, true);
  layer128(we.Wt[5], we.b[5], r1, 392, 128,   r0, 392, nullptr, 0, false);
  // gather ne[src] -> r1, ne[dst] -> r2
  {
    int row = tid >> 1, half = tid & 1, e = tileBase + row;
    int s = edges[(size_t)b * 2 * Ee + e];
    int d = edges[(size_t)b * 2 * Ee + Ee + e];
    const unsigned short* ps = ne + ((size_t)b * Nn + s) * WIDTH + half * 64;
    const unsigned short* pd = ne + ((size_t)b * Nn + d) * WIDTH + half * 64;
#pragma unroll
    for (int i = 0; i < 8; ++i) {
      *(uint4*)(r1 + row * 392 + half * 64 + i * 8) = *(const uint4*)(ps + i * 8);
      *(uint4*)(r2 + row * 392 + half * 64 + i * 8) = *(const uint4*)(pd + i * 8);
    }
  }
  __syncthreads();
  // edge_processor: first layer consumes full 384-wide concat
  layer128(wp.Wt[0], wp.b[0], bufA, 392, 384, bufB, 136, nullptr, 0, true);
  layer128(wp.Wt[1], wp.b[1], bufB, 136, 128, r1, 392, nullptr, 0, true);
  layer128(wp.Wt[2], wp.b[2], r1, 392, 128,   bufB, 136, bufB, 136, true);
  layer128(wp.Wt[3], wp.b[3], bufB, 136, 128, r1, 392, nullptr, 0, true);
  layer128(wp.Wt[4], wp.b[4], r1, 392, 128,   bufB, 136, bufB, 136, true);
  // last layer -> registers -> atomic scatter segment-sum into agg[b, src, :]
  {
    const int rowBase = (tid >> 5) * 16;
    v8f acc[8];
    gemm_rows16<8>(wp.Wt[5], 128, wp.b[5], WIDTH, bufB, 136, rowBase, acc);
    const int lane = tid & 31, g = lane >> 4, lm = lane & 15;
    for (int t = 0; t < 8; ++t) {
      int col = t * 16 + lm;
#pragma unroll
      for (int r = 0; r < 8; ++r) {
        int row = rowBase + 8 * g + r;
        int node = sIdx[row];
        atomicAdd(&agg[((size_t)b * Nn + node) * WIDTH + col], acc[t][r]);
      }
    }
  }
}

// ---- Kernel 3: fused node_processor + decoder + residual output ----
__global__ void __launch_bounds__(128) node_kernel(
    const float* __restrict__ nf, const unsigned short* __restrict__ ne,
    const float* __restrict__ agg, float* __restrict__ out,
    MlpW wn, MlpW wd, int Nn)
{
  __shared__ alignas(16) unsigned short bufA[64 * 264];   // 256-wide concat buffer
  __shared__ alignas(16) unsigned short bufB[64 * 136];
  const int b = blockIdx.y, tid = threadIdx.x;
  const int tileBase = blockIdx.x * 64;
  unsigned short* A0 = bufA;
  unsigned short* A1 = bufA + 128;
  {
    int row = tid >> 1, half = tid & 1, gr = tileBase + row;
    unsigned short* d0 = bufA + row * 264 + half * 64;
    unsigned short* d1 = bufA + row * 264 + 128 + half * 64;
    if (gr < Nn) {
      const unsigned short* ps = ne + ((size_t)b * Nn + gr) * WIDTH + half * 64;
#pragma unroll
      for (int i = 0; i < 8; ++i) *(uint4*)(d0 + i * 8) = *(const uint4*)(ps + i * 8);
      const float* pa = agg + ((size_t)b * Nn + gr) * WIDTH + half * 64;
      for (int k = 0; k < 64; ++k) d1[k] = f2bf(pa[k]);
    } else {
      for (int k = 0; k < 64; ++k) { d0[k] = 0; d1[k] = 0; }
    }
  }
  __syncthreads();
  // node_processor (input K=256)
  layer128(wn.Wt[0], wn.b[0], bufA, 264, 256, bufB, 136, nullptr, 0, true);
  layer128(wn.Wt[1], wn.b[1], bufB, 136, 128, A0, 264, nullptr, 0, true);
  layer128(wn.Wt[2], wn.b[2], A0, 264, 128,   bufB, 136, bufB, 136, true);
  layer128(wn.Wt[3], wn.b[3], bufB, 136, 128, A0, 264, nullptr, 0, true);
  layer128(wn.Wt[4], wn.b[4], A0, 264, 128,   bufB, 136, bufB, 136, true);
  layer128(wn.Wt[5], wn.b[5], bufB, 136, 128, A0, 264, nullptr, 0, false);  // npc
  // decoder
  layer128(wd.Wt[0], wd.b[0], A0, 264, 128,   bufB, 136, nullptr, 0, true);
  layer128(wd.Wt[1], wd.b[1], bufB, 136, 128, A1, 264, nullptr, 0, true);
  layer128(wd.Wt[2], wd.b[2], A1, 264, 128,   bufB, 136, bufB, 136, true);
  layer128(wd.Wt[3], wd.b[3], bufB, 136, 128, A1, 264, nullptr, 0, true);
  layer128(wd.Wt[4], wd.b[4], A1, 264, 128,   bufB, 136, bufB, 136, true);
  // last decoder layer (out width 8) + residual nf -> d_out
  {
    const int rowBase = (tid >> 5) * 16;
    v8f acc[1];
    gemm_rows16<1>(wd.Wt[5], 128, wd.b[5], 8, bufB, 136, rowBase, acc);
    const int lane = tid & 31, g = lane >> 4, lm = lane & 15;
    if (lm < 8) {
#pragma unroll
      for (int r = 0; r < 8; ++r) {
        int row = rowBase + 8 * g + r, gr = tileBase + row;
        if (gr < Nn) {
          size_t o = ((size_t)b * Nn + gr) * 8 + lm;
          out[o] = nf[o] + acc[0][r];
        }
      }
    }
  }
}

extern "C" void kernel_launch(void* const* d_in, const int* in_sizes, int n_in,
                              void* d_out, int out_size, void* d_ws, size_t ws_size,
                              hipStream_t stream)
{
  const int*   edges = (const int*)d_in[0];
  const float* nf    = (const float*)d_in[1];
  const float* ef    = (const float*)d_in[2];

  // Identify the 5 MLP param groups by their first weight's element count.
  struct Grp { const float* W[6]; const float* b[6]; };
  Grp g_ne{}, g_ee{}, g_ep{}, g_np{}, g_de{};
  int base = 3;
  for (int gi = 0; gi < 5 && base + 11 < n_in; ++gi) {
    int s = in_sizes[base];
    Grp* t = (s == 8 * WIDTH) ? &g_ne
           : (s == 4 * WIDTH) ? &g_ee
           : (s == 3 * WIDTH * WIDTH) ? &g_ep
           : (s == 2 * WIDTH * WIDTH) ? &g_np : &g_de;
    for (int l = 0; l < 6; ++l) {
      t->W[l] = (const float*)d_in[base + 2 * l];
      t->b[l] = (const float*)d_in[base + 2 * l + 1];
    }
    base += 12;
  }

  // Workspace: transposed bf16 weights | ne bf16 | agg f32
  unsigned short* wt = (unsigned short*)d_ws;
  size_t off = 0;
  struct LD { int K, Kp, N, Np; };
  const LD mid{128, 128, 128, 128};
  LD dims_ne[6] = {{8,   32, 128, 128}, mid, mid, mid, mid, mid};
  LD dims_ee[6] = {{4,   32, 128, 128}, mid, mid, mid, mid, mid};
  LD dims_ep[6] = {{384, 384,128, 128}, mid, mid, mid, mid, mid};
  LD dims_np[6] = {{256, 256,128, 128}, mid, mid, mid, mid, mid};
  LD dims_de[6] = {mid, mid, mid, mid, mid, {128, 128, 8, 16}};
  MlpW m_ne, m_ee, m_ep, m_np, m_de;
  auto setup = [&](const Grp& g, const LD* d, MlpW& m) {
    for (int l = 0; l < 6; ++l) {
      unsigned short* dst = wt + off;
      m.Wt[l] = dst; m.b[l] = g.b[l];
      int tot = d[l].Np * d[l].Kp;
      hipLaunchKernelGGL(convW, dim3((tot + 255) / 256), dim3(256), 0, stream,
                         g.W[l], dst, d[l].K, d[l].Kp, d[l].N, d[l].Np);
      off += (size_t)tot;
    }
  };
  setup(g_ne, dims_ne, m_ne);
  setup(g_ee, dims_ee, m_ee);
  setup(g_ep, dims_ep, m_ep);
  setup(g_np, dims_np, m_np);
  setup(g_de, dims_de, m_de);

  size_t wtBytes = (off * 2 + 255) & ~(size_t)255;
  unsigned short* neBuf = (unsigned short*)((char*)d_ws + wtBytes);
  size_t neBytes = ((size_t)BATCH * NNODE * WIDTH * 2 + 255) & ~(size_t)255;
  float* agg = (float*)((char*)d_ws + wtBytes + neBytes);
  hipMemsetAsync(agg, 0, (size_t)BATCH * NNODE * WIDTH * sizeof(float), stream);

  dim3 gN((NNODE + 63) / 64, BATCH), bN(128);
  dim3 gE(NEDGE / 32, BATCH), bE(64);
  hipLaunchKernelGGL(node_encoder_kernel, gN, bN, 0, stream, nf, neBuf, m_ne, NNODE);
  hipLaunchKernelGGL(edge_kernel, gE, bE, 0, stream,
                     edges, ef, neBuf, agg, m_ee, m_ep, NNODE, NEDGE);
  hipLaunchKernelGGL(node_kernel, gN, bN, 0, stream,
                     nf, neBuf, agg, (float*)d_out, m_np, m_de, NNODE);
  (void)out_size; (void)ws_size;
}
